// AttentionLayer_46677704572976
// MI455X (gfx1250) — matrix operand order repaired and checked by reading, not compiled, not run on hardware
//
#include <hip/hip_runtime.h>
#include <hip/hip_bf16.h>

// ---------------------------------------------------------------------------
// AttentionLayer, MI455X-optimized.
//
// Key fact: softmax over a singleton axis -> A == ones(N,N), so
//   out[n,:] = colsum(V) = wV @ (row-sums of kv)   for every n.
// Everything else (y, w_q1, wQ, wK, QK^T, A@V) is dead code.
// Remaining work is ~46 MB of HBM traffic (~2 us at 23.3 TB/s) plus one
// 512x512 GEMV done with v_wmma_f32_16x16x32_f16.
// ---------------------------------------------------------------------------

#define DD     512
#define IMG    96
#define NPOS   (IMG * IMG)        // 9216 = 18 * 512
#define HIMG   48
#define HSQ    (HIMG * HIMG)      // 2304

typedef __attribute__((ext_vector_type(16))) _Float16 v16h;
typedef __attribute__((ext_vector_type(8)))  float    v8f;

// workspace layout (floats):
//  [0    ,1024) u       : column sums of w_kv (1024)
//  [1024 ,2048) v_up    : w_up folded with u[512:1024]  (256 x 4)
//  [2048 ,2560) s       : row-sums of kv (512), accumulated
//  [2560 ,3072) colsumV : wV @ s (512)

__global__ void zero_ws_kernel(float* __restrict__ ws) {
    int i = blockIdx.x * blockDim.x + threadIdx.x;
    if (i < 3072) ws[i] = 0.0f;
}

// u[c'] = sum_d w_kv[d, c'],  w_kv is (512, 1024) row-major
__global__ void colsum_wkv_kernel(const float* __restrict__ w_kv,
                                  float* __restrict__ u) {
    int c = blockIdx.x * blockDim.x + threadIdx.x;   // 0..1023
    if (c >= 1024) return;
    float acc = 0.0f;
#pragma unroll 8
    for (int d = 0; d < DD; ++d) acc += w_kv[d * 1024 + c];
    u[c] = acc;
}

// v_up[i*4+ab] = sum_o u[512+o] * w_up[i, o, ab]   (w_up: (256,512,2,2))
__global__ void fold_wup_kernel(const float* __restrict__ w_up,
                                const float* __restrict__ u,
                                float* __restrict__ v_up) {
    int t = blockIdx.x * blockDim.x + threadIdx.x;   // (i, ab), 0..1023
    if (t >= 1024) return;
    int i = t >> 2, ab = t & 3;
    const float* ua = u + 512;
    const float* wp = w_up + i * 2048 + ab;
    float acc = 0.0f;
#pragma unroll 4
    for (int o = 0; o < 512; ++o) acc += ua[o] * wp[o * 4];
    v_up[t] = acc;
}

// s[c] += u[c'] * sum_{j=0..17} x[c'*9216 + j*512 + c]
// one block per input channel c' (contiguous 36 KB row), coalesced.
__global__ void reduce_x_kernel(const float* __restrict__ x,
                                const float* __restrict__ u,
                                float* __restrict__ s) {
    int cp = blockIdx.x;              // channel c' 0..511 (uniform -> SGPR)
    int c  = threadIdx.x;             // column 0..511
    const float* row = x + (size_t)cp * NPOS;
    float acc = 0.0f;
#pragma unroll
    for (int j = 0; j < 18; ++j) {
        if (j + 4 < 18) __builtin_prefetch(row + (j + 4) * 512 + c, 0, 1);
        acc += row[j * 512 + c];
    }
    atomicAdd(&s[c], acc * u[cp]);
}

// s[c] += sum over positions p == c (mod 512) of dot(ag[:,h/2,w/2], v_up[:,h%2,w%2])
__global__ void reduce_ag_kernel(const float* __restrict__ ag,
                                 const float* __restrict__ v_up,
                                 float* __restrict__ s) {
    int p = blockIdx.x * blockDim.x + threadIdx.x;   // 0..9215
    if (p >= NPOS) return;
    int h = p / IMG, w = p - h * IMG;
    int ab = ((h & 1) << 1) | (w & 1);
    const float* cell = ag + (h >> 1) * HIMG + (w >> 1);
    const float* vw = v_up + ab;
    float acc = 0.0f;
#pragma unroll 4
    for (int i = 0; i < 256; ++i) acc += cell[i * HSQ] * vw[i * 4];
    atomicAdd(&s[p & 511], acc);
}

// s[c] += sum_r pe[r, c]   (interleaved sin/cos positional encoding)
// grid (2, 32) x 256: r-range split 32 ways to parallelize the trig sum.
__global__ void pe_colsum_kernel(float* __restrict__ s) {
    int c = blockIdx.x * blockDim.x + threadIdx.x;   // 0..511
    if (c >= DD) return;
    const int chunk = NPOS / 32;                     // 288
    int r0 = blockIdx.y * chunk;
    float f = powf(1e-4f, (float)(c >> 1) * (1.0f / 256.0f));
    int odd = c & 1;
    float acc = 0.0f;
    for (int r = r0; r < r0 + chunk; ++r) {
        float a = (float)r * f;
        acc += odd ? sinf(a) : cosf(a);
    }
    atomicAdd(&s[c], acc);
}

// colsumV = wV @ s via v_wmma_f32_16x16x32_f16.
// One wave (32 threads) per 16-row tile of wV; B = s-chunk broadcast across
// all 16 N-columns (all output columns identical; lane n==0 extracts).
// s staged in LDS as f16 scaled by 2^-6 (|s| can reach ~9e3 from pesum).
__global__ void gemv_wmma_kernel(const float* __restrict__ wV,
                                 const float* __restrict__ s,
                                 float* __restrict__ colsumV) {
    __shared__ _Float16 s_lds[DD];
    int lane = threadIdx.x;                          // 0..31, one wave32
#pragma unroll
    for (int t = 0; t < 16; ++t) {
        int idx = lane * 16 + t;
        s_lds[idx] = (_Float16)(s[idx] * 0.015625f); // * 2^-6
    }
    __syncthreads();

    int tile = blockIdx.x;                           // 0..31
    int hi   = lane >> 4;                            // half-wave select
    int mrow = tile * 16 + (lane & 15);
    const float* arow = wV + (size_t)mrow * DD;
    int abase = hi ? 16 : 0;                         // 16-bit A: K base per half
    int bbase = hi ? 16 : 0;                         // 16-bit B: K base per half

    v8f acc = {};
#pragma unroll 2
    for (int kk = 0; kk < DD / 32; ++kk) {           // 16 chained WMMAs
        int k0 = kk * 32;
        v16h a, b;
#pragma unroll
        for (int e = 0; e < 16; ++e) {
            int ka = abase + e;                      // contiguous K, matches B
            a[e] = (_Float16)arow[k0 + ka];
            b[e] = s_lds[k0 + bbase + e];            // contiguous K, same order as A
        }
        acc = __builtin_amdgcn_wmma_f32_16x16x32_f16(
            /*neg_a=*/false, a, /*neg_b=*/false, b,
            /*c_mod=*/(short)0, acc, /*reuse_a=*/false, /*reuse_b=*/false);
    }

    if ((lane & 15) == 0) {                          // column N=0 holds result
        int mb = tile * 16 + (hi ? 8 : 0);
#pragma unroll
        for (int r = 0; r < 8; ++r)
            colsumV[mb + r] = acc[r] * 64.0f;        // undo 2^-6 scale
    }
}

// out_flat[k] = colsumV[k & 511]  (A == ones -> every token row identical)
__global__ void broadcast_out_kernel(const float* __restrict__ colsumV,
                                     float* __restrict__ out, int n4, int n) {
    int i = blockIdx.x * blockDim.x + threadIdx.x;
    int stride = gridDim.x * blockDim.x;
    for (; i < n4; i += stride) {
        int base = i * 4;                            // 512 % 4 == 0 -> safe
        float4 v = *(const float4*)(colsumV + (base & 511));
        *(float4*)(out + base) = v;
    }
    if (blockIdx.x == 0 && threadIdx.x == 0)
        for (int k = n4 * 4; k < n; ++k) out[k] = colsumV[k & 511];
}

extern "C" void kernel_launch(void* const* d_in, const int* in_sizes, int n_in,
                              void* d_out, int out_size, void* d_ws, size_t ws_size,
                              hipStream_t stream) {
    const float* x    = (const float*)d_in[0];
    // d_in[1] (y), d_in[5] (w_q1), d_in[6] (wQ), d_in[7] (wK): provably unused.
    const float* ag   = (const float*)d_in[2];
    const float* w_up = (const float*)d_in[3];
    const float* w_kv = (const float*)d_in[4];
    const float* wV   = (const float*)d_in[8];
    float* out = (float*)d_out;

    float* ws      = (float*)d_ws;
    float* u       = ws;
    float* v_up    = ws + 1024;
    float* s       = ws + 2048;
    float* colsumV = ws + 2560;

    zero_ws_kernel    <<<12, 256, 0, stream>>>(ws);
    colsum_wkv_kernel <<<4, 256, 0, stream>>>(w_kv, u);
    fold_wup_kernel   <<<4, 256, 0, stream>>>(w_up, u, v_up);
    reduce_x_kernel   <<<512, 512, 0, stream>>>(x, u, s);
    reduce_ag_kernel  <<<(NPOS + 255) / 256, 256, 0, stream>>>(ag, v_up, s);
    pe_colsum_kernel  <<<dim3(2, 32), 256, 0, stream>>>(s);
    gemv_wmma_kernel  <<<32, 32, 0, stream>>>(wV, s, colsumV);

    int n4 = out_size / 4;
    broadcast_out_kernel<<<1024, 256, 0, stream>>>(colsumV, out, n4, out_size);
}
